// SpikeSSM_18013092839791
// MI455X (gfx1250) — matrix operand order repaired and checked by reading, not compiled, not run on hardware
//
#include <hip/hip_runtime.h>
#include <hip/hip_bf16.h>

// ---------------------------------------------------------------------------
// Spiking SSM for MI455X (gfx1250), wave32 + WMMA.
//   B=16, S=4096, H=512.
//   Phase 0: pack Wi/Wh/Wo (f32 [K][N]) into f16 WMMA B-fragment layout.
//   Phase 1: ic_all = x @ Wi + bi   (f16 WMMA, f32 accum) -> staged in d_out.
//   Phase 2: persistent single-workgroup scan, 32 waves, Wh tile resident in
//            VGPRs (128 VGPRs/wave), h state double-buffered in LDS,
//            16 wmma (4 independent accumulator chains) + 1 barrier per step.
//   Phase 3: outputs = spikes @ Wo + bo, overwriting the ic staging in d_out.
// d_ws layout (f16): WiP(512KB) WhP(512KB) WoP(512KB) spikes(64MB).
// ---------------------------------------------------------------------------

#define HSZ   512
#define SEQ   4096
#define BATCH 16
#define NKT   (HSZ / 32)   // K-tiles of 32 per WMMA chain = 16

typedef _Float16 v16h __attribute__((ext_vector_type(16)));
typedef _Float16 v8h  __attribute__((ext_vector_type(8)));
typedef float    v8f  __attribute__((ext_vector_type(8)));

// ---------------------------------------------------------------------------
// Pack W (f32, row-major [K=512][N=512]) into per-wave B fragments:
// fragment (nt, kt) = 32 lanes x 16 contiguous halves:
//   lane = (n % 16) + 16*((kr / 8) % 2), element e = kr % 8 + 8*(kr >= 16),  kr = k % 32.
// Matches V_WMMA 16x16x32 f16 B-matrix layout (lanes 0-15: K={0..7,16..23},
// lanes 16-31: K={8..15,24..31}).
// ---------------------------------------------------------------------------
__global__ __launch_bounds__(256) void pack_w_kernel(const float* __restrict__ W,
                                                     _Float16* __restrict__ P) {
    int idx = blockIdx.x * 256 + threadIdx.x;      // 0 .. H*H-1
    int k = idx >> 9;
    int n = idx & (HSZ - 1);
    int kt = k >> 5, kr = k & 31;
    int nt = n >> 4, nc = n & 15;
    int lane = nc + ((kr & 8) << 1);
    int e    = (kr & 7) | ((kr & 16) >> 1);
    P[(((size_t)nt * NKT + kt) * 32 + lane) * 16 + e] = (_Float16)W[idx];
}

// A-fragment (16-bit A 16x32 layout): lane<16 -> K {0..7,16..23},
// lane>=16 -> K {8..15,24..31}; handled via khalf = (lane>>4)*8.

// ---------------------------------------------------------------------------
// Phase 1: ic = x @ Wi + bi.  x flat: [B*S][H] row-major. Grid (4096 Mtiles, 4),
// 8 waves/block, wave nt = blockIdx.y*8 + wave. f32 loads converted to f16.
// ---------------------------------------------------------------------------
__global__ __launch_bounds__(256) void ic_gemm_kernel(const float* __restrict__ x,
                                                      const _Float16* __restrict__ WiP,
                                                      const float* __restrict__ bi,
                                                      float* __restrict__ ic) {
    int wave = threadIdx.x >> 5, lane = threadIdx.x & 31;
    int mrow = blockIdx.x * 16;
    int nt   = blockIdx.y * 8 + wave;
    int r     = mrow + (lane & 15);
    int khalf = (lane >> 4) * 8;
    int col   = nt * 16 + (lane & 15);

    v8f acc[4] = {};   // 4 independent accumulator chains for XDL overlap
#pragma unroll
    for (int kt = 0; kt < NKT; ++kt) {
        const float* ar = x + (size_t)r * HSZ + kt * 32 + khalf;
        __builtin_prefetch(ar + 64, 0, 0);   // global_prefetch for streaming x
        v16h a;
#pragma unroll
        for (int e = 0; e < 8; ++e) {
            a[e]     = (_Float16)ar[e];
            a[e + 8] = (_Float16)ar[e + 16];
        }
        v16h bmat = *(const v16h*)(WiP + (((size_t)nt * NKT + kt) * 32 + lane) * 16);
        acc[kt & 3] = __builtin_amdgcn_wmma_f32_16x16x32_f16(
            false, a, false, bmat, (short)0, acc[kt & 3], false, false);
    }
    v8f c = (acc[0] + acc[1]) + (acc[2] + acc[3]);
    float biv = bi[col];
    int m0 = (lane >> 4) * 8;
#pragma unroll
    for (int j = 0; j < 8; ++j)
        ic[(size_t)(mrow + m0 + j) * HSZ + col] = c[j] + biv;
}

// ---------------------------------------------------------------------------
// Phase 2: persistent scan. One workgroup, 32 waves (1024 thr). Wave nt owns
// hc columns [nt*16, nt*16+16). Wh fragments pinned in VGPRs. h state
// double-buffered in LDS -> exactly one barrier per timestep.
// ---------------------------------------------------------------------------
__global__ __launch_bounds__(1024, 1) void scan_kernel(const float* __restrict__ ic,
                                                       const _Float16* __restrict__ WhP,
                                                       const float* __restrict__ bh,
                                                       _Float16* __restrict__ spk,
                                                       float* __restrict__ tail) {
    __shared__ _Float16 hbuf[2][BATCH * HSZ];   // 2 x 16 KB ping-pong

    int tid = threadIdx.x;
    int wave = tid >> 5, lane = tid & 31;
    int nt    = wave;
    int col   = nt * 16 + (lane & 15);
    int m0    = (lane >> 4) * 8;
    int khalf = (lane >> 4) * 8;
    int r     = lane & 15;

#pragma unroll
    for (int i = 0; i < 8; ++i) {
        hbuf[0][tid + i * 1024] = (_Float16)0.0f;
        hbuf[1][tid + i * 1024] = (_Float16)0.0f;
    }

    // Pin this wave's Wh column-tile in VGPRs: 16 x v16h = 128 VGPRs.
    v16h whf[NKT];
#pragma unroll
    for (int kt = 0; kt < NKT; ++kt)
        whf[kt] = *(const v16h*)(WhP + (((size_t)nt * NKT + kt) * 32 + lane) * 16);

    float bhv = bh[col];
    float hreg[8], vreg[8], sreg[8];
#pragma unroll
    for (int j = 0; j < 8; ++j) { hreg[j] = 0.f; vreg[j] = 0.f; sreg[j] = 0.f; }

    const float inv_tau = 0.1f;
    __syncthreads();

    int cur = 0;
    for (int t = 0; t < SEQ; ++t) {
        // Issue ic(t) loads first: global latency overlaps the WMMA chain.
        float icv[8];
#pragma unroll
        for (int j = 0; j < 8; ++j)
            icv[j] = ic[((size_t)(m0 + j) * SEQ + t) * HSZ + col];

        // hc_tile = h(t-1) @ Wh, 4 independent accumulator chains.
        const _Float16* hrd = &hbuf[cur][0];
        v8f acc[4] = {};
#pragma unroll
        for (int kt = 0; kt < NKT; ++kt) {
            const _Float16* hp = hrd + r * HSZ + kt * 32 + khalf;
            v8h lo = *(const v8h*)hp;         // ds_load_b128
            v8h hi = *(const v8h*)(hp + 16);  // ds_load_b128
            v16h a;
#pragma unroll
            for (int e = 0; e < 8; ++e) { a[e] = lo[e]; a[e + 8] = hi[e]; }
            acc[kt & 3] = __builtin_amdgcn_wmma_f32_16x16x32_f16(
                false, a, false, whf[kt], (short)0, acc[kt & 3], false, false);
        }
        v8f c = (acc[0] + acc[1]) + (acc[2] + acc[3]);

        // Pointwise update; write h(t) into the OTHER buffer (no WAR barrier).
        _Float16* hwr = &hbuf[cur ^ 1][0];
#pragma unroll
        for (int j = 0; j < 8; ++j) {
            int m = m0 + j;
            float hn = hreg[j] + (icv[j] + c[j] + bhv - hreg[j]) * inv_tau;
            float vn = vreg[j] + (hn - vreg[j]) * inv_tau;
            float sp = (vn >= 1.0f) ? 1.0f : 0.0f;
            vreg[j] = (sp > 0.0f) ? 0.0f : vn;
            hreg[j] = hn;
            sreg[j] = sp;
            hwr[m * HSZ + col] = (_Float16)hn;
            spk[((size_t)t * BATCH + m) * HSZ + col] = (_Float16)sp;
        }
        __syncthreads();   // h(t) visible to all waves before step t+1 reads
        cur ^= 1;
    }

    // tail = d_out + B*S*H : h_f [B,H], v_f [B,H], spikes_last [B,H]
#pragma unroll
    for (int j = 0; j < 8; ++j) {
        int m = m0 + j;
        tail[(size_t)m * HSZ + col]               = hreg[j];
        tail[(size_t)(BATCH + m) * HSZ + col]     = vreg[j];
        tail[(size_t)(2 * BATCH + m) * HSZ + col] = sreg[j];
    }
}

// ---------------------------------------------------------------------------
// Phase 3: outputs[b,s,d] = spikes[s,b,:] @ Wo + bo. One M-tile == one timestep
// (rows are the 16 batches). Grid (4096, 4), 8 waves/block.
// ---------------------------------------------------------------------------
__global__ __launch_bounds__(256) void out_gemm_kernel(const _Float16* __restrict__ spk,
                                                       const _Float16* __restrict__ WoP,
                                                       const float* __restrict__ bo,
                                                       float* __restrict__ out) {
    int wave = threadIdx.x >> 5, lane = threadIdx.x & 31;
    int t  = blockIdx.x;
    int nt = blockIdx.y * 8 + wave;
    int col   = nt * 16 + (lane & 15);
    int khalf = (lane >> 4) * 8;
    int r     = lane & 15;

    v8f acc[4] = {};
#pragma unroll
    for (int kt = 0; kt < NKT; ++kt) {
        const _Float16* ap = spk + ((size_t)t * BATCH + r) * HSZ + kt * 32 + khalf;
        v8h lo = *(const v8h*)ap;
        v8h hi = *(const v8h*)(ap + 16);
        v16h a;
#pragma unroll
        for (int e = 0; e < 8; ++e) { a[e] = lo[e]; a[e + 8] = hi[e]; }
        v16h bmat = *(const v16h*)(WoP + (((size_t)nt * NKT + kt) * 32 + lane) * 16);
        acc[kt & 3] = __builtin_amdgcn_wmma_f32_16x16x32_f16(
            false, a, false, bmat, (short)0, acc[kt & 3], false, false);
    }
    v8f c = (acc[0] + acc[1]) + (acc[2] + acc[3]);
    float bov = bo[col];
    int m0 = (lane >> 4) * 8;
#pragma unroll
    for (int j = 0; j < 8; ++j) {
        int bb = m0 + j;
        out[((size_t)bb * SEQ + t) * HSZ + col] = c[j] + bov;
    }
}

// ---------------------------------------------------------------------------
extern "C" void kernel_launch(void* const* d_in, const int* in_sizes, int n_in,
                              void* d_out, int out_size, void* d_ws, size_t ws_size,
                              hipStream_t stream) {
    (void)in_sizes; (void)n_in; (void)out_size; (void)ws_size;
    const float* x  = (const float*)d_in[0];
    const float* Wi = (const float*)d_in[1];
    const float* bi = (const float*)d_in[2];
    const float* Wh = (const float*)d_in[3];
    const float* bh = (const float*)d_in[4];
    const float* Wo = (const float*)d_in[5];
    const float* bo = (const float*)d_in[6];
    float* out = (float*)d_out;

    _Float16* WiP = (_Float16*)d_ws;
    _Float16* WhP = WiP + (size_t)HSZ * HSZ;
    _Float16* WoP = WhP + (size_t)HSZ * HSZ;
    _Float16* spk = WoP + (size_t)HSZ * HSZ;          // S*B*H f16 (64 MB)

    float* ic   = out;                                 // stage ic_all in d_out
    float* tail = out + (size_t)BATCH * SEQ * HSZ;     // h_f, v_f, spikes_last

    pack_w_kernel<<<dim3(HSZ * HSZ / 256), 256, 0, stream>>>(Wi, WiP);
    pack_w_kernel<<<dim3(HSZ * HSZ / 256), 256, 0, stream>>>(Wh, WhP);
    pack_w_kernel<<<dim3(HSZ * HSZ / 256), 256, 0, stream>>>(Wo, WoP);

    ic_gemm_kernel<<<dim3((BATCH * SEQ) / 16, 4), 256, 0, stream>>>(x, WiP, bi, ic);
    scan_kernel<<<dim3(1), 1024, 0, stream>>>(ic, WhP, bh, spk, tail);
    out_gemm_kernel<<<dim3(SEQ, 4), 256, 0, stream>>>(spk, WoP, bo, out);
}
